// KinematicsEncoder_41979010351511
// MI455X (gfx1250) — matrix operand order repaired and checked by reading, not compiled, not run on hardware
//
#include <hip/hip_runtime.h>
#include <hip/hip_bf16.h>
#include <cstdint>

typedef _Float16 v16h __attribute__((ext_vector_type(16)));
typedef _Float16 v8h  __attribute__((ext_vector_type(8)));
typedef float    v8f  __attribute__((ext_vector_type(8)));

constexpr int NB  = 1024;   // batch
constexpr int NT  = 128;    // seq len
constexpr int NI  = 64;     // input size
constexpr int NH  = 256;    // hidden
constexpr int NG  = 4*NH;   // 1024 = 4H (gates)
constexpr int K2H = 2*NH;   // 512  = [h;c] width

constexpr float LOG2E = 1.4426950408889634f;

// async ai chunking: 2 i-columns per chunk -> 32 B128 issues/chunk (fits 6-bit ASYNCcnt)
constexpr int CH       = 2;            // i per chunk
constexpr int NCHUNK   = NI / CH;      // 32 chunks per step
constexpr int AIPITCH  = 132;          // padded row stride (floats) to break bank conflicts

__device__ __forceinline__ float fexp2(float x){
#if __has_builtin(__builtin_amdgcn_exp2f)
  return __builtin_amdgcn_exp2f(x);
#else
  return exp2f(x);
#endif
}
__device__ __forceinline__ float frcp(float x){
#if __has_builtin(__builtin_amdgcn_rcpf)
  return __builtin_amdgcn_rcpf(x);
#else
  return 1.0f/x;
#endif
}
__device__ __forceinline__ float ftanh(float x){
#if __has_builtin(__builtin_amdgcn_tanhf)
  return __builtin_amdgcn_tanhf(x);        // CDNA5 V_TANH_F32 (TRANS)
#else
  float e = fexp2(__builtin_fabsf(x) * (-2.0f*LOG2E));   // in (0,1]
  float r = (1.0f - e) * frcp(1.0f + e);
  return __builtin_copysignf(r, x);
#endif
}
__device__ __forceinline__ float fsigm(float x){
  return frcp(1.0f + fexp2(x * (-LOG2E)));
}

// ---- CDNA5 async global -> LDS (ASYNCcnt-tracked), inline asm (toolchain-portable) ----
__device__ __forceinline__ void async_load_b128(const float* gptr, float* lptr){
  unsigned lds = (unsigned)(uintptr_t)lptr;             // flat LDS addr low 32b = ds offset
  unsigned long long g = (unsigned long long)(uintptr_t)gptr;
  asm volatile("global_load_async_to_lds_b128 %0, %1, off"
               :: "v"(lds), "v"(g) : "memory");
}
__device__ __forceinline__ void wait_async_le32(){
  asm volatile("s_wait_asynccnt 0x20" ::: "memory");    // older chunk (in-order) complete
}
__device__ __forceinline__ void wait_async_0(){
  asm volatile("s_wait_asynccnt 0x0" ::: "memory");
}

__device__ __forceinline__ v16h cat8(v8h lo, v8h hi){
  return __builtin_shufflevector(lo, hi, 0,1,2,3,4,5,6,7,8,9,10,11,12,13,14,15);
}
// A fragment: 16xK row-major f16 in LDS, M = lane%16.
// half=0 holds K = kb*32 + {0..15}; half=1 holds {16..31} (matches b_frag).
__device__ __forceinline__ v16h a_frag(const _Float16* base, int stride, int m, int kb, int half){
  const _Float16* p = base + m*stride + kb*32 + half*16;
  return cat8(*(const v8h*)p, *(const v8h*)(p + 8));
}
// B fragment from an N x K row-major f16 weight (output-major).
// B order used here (same as a_frag): lane n = lane%16 holds column n; K = kb*32 + 16*half + {0..15} (contiguous 32B).
__device__ __forceinline__ v16h b_frag(const _Float16* W, int K, int row, int kb, int half){
  const _Float16* p = W + (size_t)row*(size_t)K + kb*32 + half*16;
  return cat8(*(const v8h*)p, *(const v8h*)(p + 8));
}
__device__ __forceinline__ v8f wmma16(v16h a, v16h b, v8f c){
  return __builtin_amdgcn_wmma_f32_16x16x32_f16(false, a, false, b, (short)0, c, false, false);
}

// ---------------- weight prep: f32 -> f16 (N x K row-major kept), bias fuse ----------------
__global__ __launch_bounds__(256) void prep_weights(
    const float* __restrict__ W_ih, const float* __restrict__ W_hh,
    const float* __restrict__ b_ih, const float* __restrict__ b_hh,
    const float* __restrict__ W_e,
    _Float16* __restrict__ Wih_h, _Float16* __restrict__ Whh_h,
    _Float16* __restrict__ We_h, float* __restrict__ bsum)
{
  int idx = blockIdx.x*blockDim.x + threadIdx.x;
  if (idx < NG*NH)  Whh_h[idx] = (_Float16)W_hh[idx];
  if (idx < NG*NI)  Wih_h[idx] = (_Float16)W_ih[idx];
  if (idx < NT*K2H) We_h[idx]  = (_Float16)W_e[idx];
  if (idx < NG)     bsum[idx]  = b_ih[idx] + b_hh[idx];
}

// ---------------- loop-invariant attention input: ai[b][i][s] = sum_t x[b][t][i] * V_e[s][t] ----
__global__ __launch_bounds__(128) void attn_input_kernel(
    const float* __restrict__ x, const float* __restrict__ V_e, float* __restrict__ ai)
{
  __shared__ float sx[NT*NI];                 // 32 KB: x[b] tile
  const int b = blockIdx.x;
  const float* xb = x + (size_t)b*NT*NI;
  for (int idx = threadIdx.x; idx < NT*NI; idx += 128) sx[idx] = xb[idx];
  __syncthreads();
  const int s = threadIdx.x;                  // 0..127
  const float* vrow = V_e + s*NT;             // contiguous per lane row, L2-resident
  for (int i = 0; i < NI; ++i) {
    float acc = 0.f;
    #pragma unroll 8
    for (int t = 0; t < NT; ++t) acc = fmaf(sx[t*NI + i], vrow[t], acc);
    ai[((size_t)b*NI + i)*NT + s] = acc;      // coalesced over s
  }
}

// ---------------- persistent recurrence: one wave owns a 16-batch tile for all T steps --------
__global__ __launch_bounds__(32) void lstm_attn_persistent(
    const float* __restrict__ x, const float* __restrict__ ai,
    const _Float16* __restrict__ Wih_h, const _Float16* __restrict__ Whh_h,
    const _Float16* __restrict__ We_h, const float* __restrict__ bsum,
    const float* __restrict__ ve,
    float* __restrict__ out_hidden, float* __restrict__ out_h, float* __restrict__ out_c)
{
  __shared__ __align__(16) _Float16 s_hc[16*K2H];        // f16 [h|c], WMMA A operand (16 KB)
  __shared__ __align__(16) float    s_c [16*NH];         // f32 cell state            (16 KB)
  __shared__ __align__(16) float    s_h [16*NH];         // f32 hidden                (16 KB)
  __shared__ __align__(16) float    s_ah[16*NT];         // attn_hidden               ( 8 KB)
  __shared__ __align__(16) float    s_e [16*NI];         // energies / softmax probs  ( 4 KB)
  __shared__ __align__(16) _Float16 s_xt[16*NI];         // x_tilde f16               ( 2 KB)
  __shared__ __align__(16) float    s_ai[2][16*CH*AIPITCH]; // async ai double buffer (33 KB)
  __shared__ __align__(16) float    s_ve[NT];
  __shared__ float s_inv[16];

  const int lane = threadIdx.x;
  const int n16  = lane & 15;     // WMMA N (and A-row M) lane index
  const int half = lane >> 4;
  const int b0   = blockIdx.x * 16;

  for (int idx = lane; idx < 16*K2H; idx += 32) s_hc[idx] = (_Float16)0.f;
  for (int idx = lane; idx < 16*NH;  idx += 32) { s_c[idx] = 0.f; s_h[idx] = 0.f; }
  for (int idx = lane; idx < NT;     idx += 32) s_ve[idx] = ve[idx];
  for (int idx = lane; idx < 16*CH*AIPITCH; idx += 32) { s_ai[0][idx] = 0.f; s_ai[1][idx] = 0.f; }
  __syncthreads();

  // per-lane bases for the attention-energy phase: m = n16, s-half = half
  const float* aib = ai + ((size_t)(b0 + n16)*NI)*NT + half*64;   // + i*NT
  float* lbase0 = &s_ai[0][(n16*CH)*AIPITCH + half*64];
  float* lbase1 = &s_ai[1][(n16*CH)*AIPITCH + half*64];

  for (int t = 0; t < NT; ++t) {
    // ---- attn_hidden(16x128) = [h;c](16x512,f16) @ W_e^T : 8 N-tiles x 16 k-slabs WMMA ----
    for (int nt = 0; nt < 8; ++nt) {
      v8f acc = {};
      const int row = nt*16 + n16;
      #pragma unroll 4
      for (int kb = 0; kb < 16; ++kb) {
        v16h a = a_frag(s_hc, K2H, n16, kb, half);
        v16h b = b_frag(We_h, K2H, row, kb, half);
        acc = wmma16(a, b, acc);
      }
      #pragma unroll
      for (int r = 0; r < 8; ++r)                       // D: lanes 0-15 rows r, 16-31 rows r+8
        s_ah[(r + 8*half)*NT + nt*16 + n16] = acc[r];
    }
    __syncthreads();

    // ---- e[m][i] = sum_s ve[s]*tanh(ah[m][s] + ai[b][i][s])
    //      async double-buffered ai pipeline: chunk = CH i-columns = 32 B128 issues ----
    {
      const float* ahr = &s_ah[n16*NT + half*64];
      const float* ver = &s_ve[half*64];

      // prologue: issue chunk 0 into buffer 0
      #pragma unroll
      for (int ic = 0; ic < CH; ++ic)
        #pragma unroll
        for (int q = 0; q < 16; ++q)
          async_load_b128(aib + (size_t)ic*NT + q*4, lbase0 + ic*AIPITCH + q*4);

      for (int c = 0; c < NCHUNK; ++c) {
        float* lnext = ((c+1) & 1) ? lbase1 : lbase0;
        if (c + 1 < NCHUNK) {
          const float* gnext = aib + (size_t)(c+1)*CH*NT;
          #pragma unroll
          for (int ic = 0; ic < CH; ++ic)
            #pragma unroll
            for (int q = 0; q < 16; ++q)
              async_load_b128(gnext + (size_t)ic*NT + q*4, lnext + ic*AIPITCH + q*4);
          wait_async_le32();                  // chunk c landed (async loads complete in order)
        } else {
          wait_async_0();
        }
        const float* lcur = (c & 1) ? lbase1 : lbase0;
        #pragma unroll
        for (int ic = 0; ic < CH; ++ic) {
          const float* ap = lcur + ic*AIPITCH;
          float acc = 0.f;
          #pragma unroll 8
          for (int s = 0; s < 64; ++s)
            acc = fmaf(ver[s], ftanh(ahr[s] + ap[s]), acc);
          acc += __shfl_xor(acc, 16, 32);     // fold the two s-halves (wave32)
          s_e[n16*NI + c*CH + ic] = acc;
        }
      }
    }
    __syncthreads();

    // ---- softmax over i (64) per row m ----
    if (lane < 16) {
      const int m = lane;
      float mx = -3.4e38f;
      for (int i = 0; i < NI; ++i) mx = fmaxf(mx, s_e[m*NI + i]);
      float sum = 0.f;
      for (int i = 0; i < NI; ++i) {
        float p = fexp2((s_e[m*NI + i] - mx) * LOG2E);
        s_e[m*NI + i] = p; sum += p;
      }
      s_inv[m] = frcp(sum);
    }
    __syncthreads();
    for (int idx = lane; idx < 16*NI; idx += 32) {
      const int m = idx >> 6, i = idx & 63;
      float alpha = s_e[idx] * s_inv[m];
      float xv = x[((size_t)(b0 + m)*NT + t)*NI + i];
      s_xt[idx] = (_Float16)(alpha * xv);
    }
    __syncthreads();

    // ---- gates(16x1024) = x_tilde@W_ih^T + h@W_hh^T + b : A-frags cached in VGPRs ----
    v16h axt[2];
    #pragma unroll
    for (int kb = 0; kb < 2; ++kb) axt[kb] = a_frag(s_xt, NI, n16, kb, half);
    v16h ahh[8];
    #pragma unroll
    for (int kb = 0; kb < 8; ++kb) ahh[kb] = a_frag(s_hc, K2H, n16, kb, half); // h = cols 0..255

    for (int hb = 0; hb < 16; ++hb) {
      v8f g[4];
      #pragma unroll
      for (int gi = 0; gi < 4; ++gi) {                  // gate order i,f,g,o
        const int row = gi*256 + hb*16 + n16;
        const float bv = bsum[row];
        v8f acc;
        #pragma unroll
        for (int r = 0; r < 8; ++r) acc[r] = bv;        // bias broadcast over M
        #pragma unroll
        for (int kb = 0; kb < 2; ++kb)
          acc = wmma16(axt[kb], b_frag(Wih_h, NI, row, kb, half), acc);
        #pragma unroll
        for (int kb = 0; kb < 8; ++kb)
          acc = wmma16(ahh[kb], b_frag(Whh_h, NH, row, kb, half), acc);
        g[gi] = acc;
      }
      #pragma unroll
      for (int r = 0; r < 8; ++r) {
        const int m = r + 8*half;
        const int j = hb*16 + n16;
        float cold = s_c[m*NH + j];
        float ig = fsigm(g[0][r]);
        float fg = fsigm(g[1][r]);
        float gg = ftanh(g[2][r]);
        float og = fsigm(g[3][r]);
        float cn = fg*cold + ig*gg;
        float hn = og*ftanh(cn);
        s_c [m*NH + j]       = cn;
        s_h [m*NH + j]       = hn;
        s_hc[m*K2H + j]      = (_Float16)hn;            // h half of [h|c]
        s_hc[m*K2H + NH + j] = (_Float16)cn;            // c half of [h|c]
      }
    }
    __syncthreads();

    // ---- emit all_hidden[:, t, :] coalesced along hidden dim ----
    for (int idx = lane; idx < 16*NH; idx += 32) {
      const int m = idx >> 8, j = idx & 255;
      out_hidden[((size_t)(b0 + m)*NT + t)*NH + j] = s_h[idx];
    }
    __syncthreads();
  }

  for (int idx = lane; idx < 16*NH; idx += 32) {
    const int m = idx >> 8, j = idx & 255;
    out_h[(size_t)(b0 + m)*NH + j] = s_h[idx];
    out_c[(size_t)(b0 + m)*NH + j] = s_c[idx];
  }
}

extern "C" void kernel_launch(void* const* d_in, const int* in_sizes, int n_in,
                              void* d_out, int out_size, void* d_ws, size_t ws_size,
                              hipStream_t stream) {
  (void)in_sizes; (void)n_in; (void)out_size; (void)ws_size;
  const float* x    = (const float*)d_in[0];
  const float* W_ih = (const float*)d_in[1];
  const float* W_hh = (const float*)d_in[2];
  const float* b_ih = (const float*)d_in[3];
  const float* b_hh = (const float*)d_in[4];
  const float* W_e  = (const float*)d_in[5];
  const float* V_e  = (const float*)d_in[6];
  const float* v_e  = (const float*)d_in[7];

  // workspace layout (all offsets 256B aligned)
  char* ws = (char*)d_ws;
  float*    ai    = (float*)ws;                        // NB*NI*NT f32 = 33,554,432 B
  _Float16* We_h  = (_Float16*)(ws + 33554432);        // 128*512  f16 = 131,072 B
  _Float16* Wih_h = (_Float16*)(ws + 33685504);        // 1024*64  f16 = 131,072 B
  _Float16* Whh_h = (_Float16*)(ws + 33816576);        // 1024*256 f16 = 524,288 B
  float*    bsum  = (float*)(ws + 34340864);           // 1024 f32

  float* out        = (float*)d_out;
  float* out_hidden = out;                             // (B,T,H)
  float* out_h      = out + (size_t)NB*NT*NH;          // (B,H)
  float* out_c      = out_h + (size_t)NB*NH;           // (B,H)

  prep_weights<<<(NG*NH + 255)/256, 256, 0, stream>>>(
      W_ih, W_hh, b_ih, b_hh, W_e, Wih_h, Whh_h, We_h, bsum);
  attn_input_kernel<<<NB, 128, 0, stream>>>(x, V_e, ai);
  lstm_attn_persistent<<<NB/16, 32, 0, stream>>>(
      x, ai, Wih_h, Whh_h, We_h, bsum, v_e, out_hidden, out_h, out_c);
}
